// Model_8031588843716
// MI455X (gfx1250) — matrix operand order repaired and checked by reading, not compiled, not run on hardware
//
#include <hip/hip_runtime.h>
#include <hip/hip_bf16.h>
#include <math.h>

typedef __bf16 bf16;
typedef __attribute__((ext_vector_type(16))) __bf16 v16bf;
typedef __attribute__((ext_vector_type(8)))  float  v8f;

// ---------------------------------------------------------------- helpers
__device__ __forceinline__ bf16 f2bf(float f) {
  union { float f; unsigned int u; } in; in.f = f;
  unsigned int u = in.u;
  unsigned int r = (u + 0x7fffu + ((u >> 16) & 1u)) >> 16;   // RNE
  union { unsigned short s; bf16 b; } out; out.s = (unsigned short)r;
  return out.b;
}

__global__ void f2bf_kernel(const float* __restrict__ s, bf16* __restrict__ d, long n) {
  long e = (long)blockIdx.x * blockDim.x + threadIdx.x;
  if (e < n) d[e] = f2bf(s[e]);
}

__global__ void fill_f32_kernel(float* p, float v, long n) {
  long e = (long)blockIdx.x * blockDim.x + threadIdx.x;
  if (e < n) p[e] = v;
}

__global__ void fill_u16_kernel(unsigned short* p, long n) {
  long e = (long)blockIdx.x * blockDim.x + threadIdx.x;
  if (e < n) p[e] = 0;
}

// ---------------------------------------------------------------- GEMM
// C[b] = A[b](MxK, bf16 row-major) x W[b](NxK, bf16 row-major)^T  (+bias +resid, gelu)
// M,N multiples of 128; K multiple of 32. Outputs: optional f32 and/or bf16.
__global__ __launch_bounds__(256) void gemm_bf16_kernel(
    const bf16* __restrict__ A, long sAb,
    const bf16* __restrict__ W, long sWb,
    const float* __restrict__ bias,
    const float* __restrict__ resid, long sRb,
    float* __restrict__ Cf, long sCf,
    bf16* __restrict__ Cb, long sCbb,
    int M, int N, int K, int act)
{
  __shared__ unsigned int sA[128][17];
  __shared__ unsigned int sB[128][17];
  const int b = blockIdx.z;
  const unsigned int* Au = (const unsigned int*)(A + (long)b * sAb);
  const unsigned int* Wu = (const unsigned int*)(W + (long)b * sWb);
  const int tile_n = blockIdx.x * 128;
  const int tile_m = blockIdx.y * 128;
  const int tid  = threadIdx.x;
  const int lane = tid & 31;
  const int wave = tid >> 5;
  const int wm   = (wave >> 1) << 5;   // 0,32,64,96
  const int wn   = (wave & 1) << 6;    // 0,64
  const int l15  = lane & 15;
  const int hi   = lane >> 4;
  const int Kd2  = K >> 1;
  const int lrow = tid >> 1;           // 0..127
  const int lcol = (tid & 1) << 3;     // packed-uint col: 0 or 8

  v8f acc[2][4];
  #pragma unroll
  for (int i = 0; i < 2; ++i)
    #pragma unroll
    for (int j = 0; j < 4; ++j)
      #pragma unroll
      for (int r = 0; r < 8; ++r) acc[i][j][r] = 0.0f;

  const unsigned int* gA = Au + (size_t)(tile_m + lrow) * Kd2 + lcol;
  const unsigned int* gB = Wu + (size_t)(tile_n + lrow) * Kd2 + lcol;

  for (int k0 = 0; k0 < K; k0 += 32) {
    const uint4* pa = (const uint4*)(gA + (k0 >> 1));
    const uint4* pb = (const uint4*)(gB + (k0 >> 1));
    uint4 a0 = pa[0], a1 = pa[1];
    uint4 b0 = pb[0], b1 = pb[1];
    if (k0 + 32 < K) {                 // hint next K tile -> global_prefetch_b8
      __builtin_prefetch(gA + ((k0 + 32) >> 1), 0, 1);
      __builtin_prefetch(gB + ((k0 + 32) >> 1), 0, 1);
    }
    sA[lrow][lcol+0]=a0.x; sA[lrow][lcol+1]=a0.y; sA[lrow][lcol+2]=a0.z; sA[lrow][lcol+3]=a0.w;
    sA[lrow][lcol+4]=a1.x; sA[lrow][lcol+5]=a1.y; sA[lrow][lcol+6]=a1.z; sA[lrow][lcol+7]=a1.w;
    sB[lrow][lcol+0]=b0.x; sB[lrow][lcol+1]=b0.y; sB[lrow][lcol+2]=b0.z; sB[lrow][lcol+3]=b0.w;
    sB[lrow][lcol+4]=b1.x; sB[lrow][lcol+5]=b1.y; sB[lrow][lcol+6]=b1.z; sB[lrow][lcol+7]=b1.w;
    __syncthreads();

    union { v16bf v; unsigned int u[8]; } af[2], bfg[4];
    #pragma unroll
    for (int i = 0; i < 2; ++i) {      // A 16x32 frag: contiguous k = 16*hi + s per lane
      int row = wm + (i << 4) + l15;
      #pragma unroll
      for (int v = 0; v < 8; ++v) {
        int p = v + (hi << 3);
        af[i].u[v] = sA[row][p];
      }
    }
    #pragma unroll
    for (int j = 0; j < 4; ++j) {      // B 32x16 frag
      int col = wn + (j << 4) + l15;
      #pragma unroll
      for (int v = 0; v < 8; ++v) {
        int p = v + (hi << 3);
        bfg[j].u[v] = sB[col][p];
      }
    }
    #pragma unroll
    for (int i = 0; i < 2; ++i)
      #pragma unroll
      for (int j = 0; j < 4; ++j)
        acc[i][j] = __builtin_amdgcn_wmma_f32_16x16x32_bf16(
            false, af[i].v, false, bfg[j].v, (short)0, acc[i][j], false, false);
    __syncthreads();
  }

  const float* rp  = resid ? resid + (long)b * sRb : nullptr;
  float*       cfp = Cf    ? Cf    + (long)b * sCf : nullptr;
  bf16*        cbp = Cb    ? Cb    + (long)b * sCbb: nullptr;
  #pragma unroll
  for (int i = 0; i < 2; ++i) {
    #pragma unroll
    for (int j = 0; j < 4; ++j) {
      int col = tile_n + wn + (j << 4) + l15;
      #pragma unroll
      for (int r = 0; r < 8; ++r) {
        int row = tile_m + wm + (i << 4) + (hi << 3) + r;
        float x = acc[i][j][r];
        if (bias) x += bias[col];
        if (rp)   x += rp[(size_t)row * N + col];
        if (act == 1) x = 0.5f * x * (1.0f + erff(x * 0.70710678118654752f)); // exact gelu
        size_t idx = (size_t)row * N + col;
        if (cfp) cfp[idx] = x;
        if (cbp) cbp[idx] = f2bf(x);
      }
    }
  }
}

// ---------------------------------------------------------------- embeddings
// out[b,t,o] = sum_{kk,c} x[b,(t+kk-1)%L,c]*W[o,c,kk] + time(b,t)*temb[o]
__global__ __launch_bounds__(256) void conv_emb_kernel(
    const float* __restrict__ xin, const float* __restrict__ Wc,
    const float* __restrict__ t1, const float* __restrict__ t2, int Lsplit,
    const float* __restrict__ temb,
    float* __restrict__ outF, bf16* __restrict__ outB, int L, int Cin)
{
  int r = blockIdx.x;             // b*L + t
  int b = r / L, t = r % L;
  __shared__ float sx[3 * 32];
  __shared__ float stv;
  int tid = threadIdx.x;
  if (tid < 3 * Cin) {
    int kk = tid / Cin, c = tid % Cin;
    int srow = t + kk - 1; if (srow < 0) srow += L; if (srow >= L) srow -= L;
    sx[kk * Cin + c] = xin[((size_t)b * L + srow) * Cin + c];
  }
  if (tid == 0)
    stv = (t < Lsplit) ? t1[(size_t)b * Lsplit + t]
                       : t2[(size_t)b * (L - Lsplit) + (t - Lsplit)];
  __syncthreads();
  for (int o = tid; o < 512; o += 256) {
    float acc = stv * temb[o];
    const float* wr = Wc + (size_t)o * Cin * 3;
    for (int c = 0; c < Cin; ++c)
      acc += sx[c] * wr[c*3+0] + sx[Cin+c] * wr[c*3+1] + sx[2*Cin+c] * wr[c*3+2];
    size_t idx = (size_t)r * 512 + o;
    outF[idx] = acc;
    outB[idx] = f2bf(acc);
  }
}

// ---------------------------------------------------------------- series decomp (k=25, edge-replicate)
__global__ void decomp_kernel(const float* __restrict__ X, int L, int Cd,
                              float* __restrict__ seasF, bf16* __restrict__ seasB,
                              float* __restrict__ trendAcc, long n)
{
  long e = (long)blockIdx.x * blockDim.x + threadIdx.x;
  if (e >= n) return;
  int c = (int)(e % Cd);
  long rt = e / Cd;
  int t = (int)(rt % L);
  long bb = rt / L;
  const float* col = X + (bb * L) * (long)Cd + c;
  float s = 0.f;
  #pragma unroll 5
  for (int j = -12; j <= 12; ++j) {
    int tt = t + j; tt = tt < 0 ? 0 : (tt >= L ? L - 1 : tt);
    s += col[(long)tt * Cd];
  }
  float m = s * (1.0f / 25.0f);
  float se = X[e] - m;
  if (seasF)    seasF[e] = se;
  if (seasB)    seasB[e] = f2bf(se);
  if (trendAcc) trendAcc[e] += m;
}

// ---------------------------------------------------------------- correlation diag-mean
// mc[b,tau] = scale * sum_t C[b, t, (t-tau) mod L]
__global__ __launch_bounds__(256) void corr_mean_kernel(const float* __restrict__ Cm, int L,
                                                        float* __restrict__ mc, float scale)
{
  int tau = blockIdx.x, b = blockIdx.y, tid = threadIdx.x;
  const float* Cb_ = Cm + (size_t)b * L * L;
  float s = 0.f;
  for (int t = tid; t < L; t += 256) {
    int ss = t - tau; if (ss < 0) ss += L;
    s += Cb_[(size_t)t * L + ss];
  }
  __shared__ float red[256];
  red[tid] = s; __syncthreads();
  for (int st = 128; st > 0; st >>= 1) { if (tid < st) red[tid] += red[tid + st]; __syncthreads(); }
  if (tid == 0) mc[(size_t)b * L + tau] = red[0] * scale;
}

// ---------------------------------------------------------------- top-k + softmax (one block per batch)
__global__ __launch_bounds__(256) void topk_softmax_kernel(const float* __restrict__ mc, int L, int k,
                                                           int* __restrict__ dly, float* __restrict__ wts)
{
  int b = blockIdx.x, tid = threadIdx.x;
  __shared__ float sv[1536];
  __shared__ float rv[256];
  __shared__ int   ri[256];
  __shared__ float sel[64];
  __shared__ int   seli[64];
  for (int i = tid; i < L; i += 256) sv[i] = mc[(size_t)b * L + i];
  __syncthreads();
  for (int it = 0; it < k; ++it) {
    float best = -3.4e38f; int bi = 0;
    for (int i = tid; i < L; i += 256) { float v = sv[i]; if (v > best) { best = v; bi = i; } }
    rv[tid] = best; ri[tid] = bi; __syncthreads();
    for (int s = 128; s > 0; s >>= 1) {
      if (tid < s && rv[tid + s] > rv[tid]) { rv[tid] = rv[tid + s]; ri[tid] = ri[tid + s]; }
      __syncthreads();
    }
    if (tid == 0) { sel[it] = rv[0]; seli[it] = ri[0]; sv[ri[0]] = -3.4e38f; }
    __syncthreads();
  }
  if (tid == 0) {
    float mx = sel[0], s = 0.f;
    for (int i = 0; i < k; ++i) { float e = __expf(sel[i] - mx); sel[i] = e; s += e; }
    float inv = 1.0f / s;
    for (int i = 0; i < k; ++i) { wts[b * k + i] = sel[i] * inv; dly[b * k + i] = seli[i]; }
  }
}

// ---------------------------------------------------------------- delay aggregation
__global__ __launch_bounds__(256) void agg_kernel(const float* __restrict__ V,
    const int* __restrict__ dly, const float* __restrict__ wts, int k,
    bf16* __restrict__ outB, int L, long n)
{
  int b = blockIdx.y, tid = threadIdx.x;
  __shared__ int   sd[40];
  __shared__ float sw[40];
  if (tid < k) { sd[tid] = dly[b * k + tid]; sw[tid] = wts[b * k + tid]; }
  __syncthreads();
  long e = (long)blockIdx.x * 256 + tid;      // t*512 + d within batch
  if (e >= n) return;
  int d = (int)(e % 512), t = (int)(e / 512);
  const float* Vb = V + (size_t)b * L * 512;
  float s = 0.f;
  for (int i = 0; i < k; ++i) {
    int tt = t + sd[i]; if (tt >= L) tt -= L;
    s += sw[i] * Vb[(size_t)tt * 512 + d];
  }
  outB[(size_t)b * L * 512 + e] = f2bf(s);
}

// ---------------------------------------------------------------- layernorm (row) + column mean subtract
__global__ __launch_bounds__(256) void ln_row_kernel(const float* __restrict__ X,
    const float* __restrict__ g, const float* __restrict__ bb, float* __restrict__ Y)
{
  int r = blockIdx.x, tid = threadIdx.x;
  const float* xr = X + (size_t)r * 512;
  float x0 = xr[tid], x1 = xr[tid + 256];
  __shared__ float red[256];
  red[tid] = x0 + x1; __syncthreads();
  for (int s = 128; s > 0; s >>= 1) { if (tid < s) red[tid] += red[tid + s]; __syncthreads(); }
  float mu = red[0] * (1.f / 512.f);
  __syncthreads();
  float d0 = x0 - mu, d1 = x1 - mu;
  red[tid] = d0 * d0 + d1 * d1; __syncthreads();
  for (int s = 128; s > 0; s >>= 1) { if (tid < s) red[tid] += red[tid + s]; __syncthreads(); }
  float rs = rsqrtf(red[0] * (1.f / 512.f) + 1e-5f);
  float* yr = Y + (size_t)r * 512;
  yr[tid]       = d0 * rs * g[tid]       + bb[tid];
  yr[tid + 256] = d1 * rs * g[tid + 256] + bb[tid + 256];
}

__global__ void colmean_kernel(const float* __restrict__ X, int L, float* __restrict__ cm, long n)
{
  long e = (long)blockIdx.x * blockDim.x + threadIdx.x;
  if (e >= n) return;
  int d = (int)(e % 512); long b = e / 512;
  const float* p = X + (b * L) * 512L + d;
  float s = 0.f;
  for (int t = 0; t < L; ++t) s += p[(size_t)t * 512];
  cm[e] = s / (float)L;
}

__global__ void colsub_kernel(const float* __restrict__ X, const float* __restrict__ cm,
                              float* __restrict__ outF, bf16* __restrict__ outB, int L, long n)
{
  long e = (long)blockIdx.x * blockDim.x + threadIdx.x;
  if (e >= n) return;
  int d = (int)(e % 512);
  long b = e / ((long)L * 512);
  float v = X[e] - cm[b * 512 + d];
  outF[e] = v;
  if (outB) outB[e] = f2bf(v);
}

// ---------------------------------------------------------------- small decoder helpers
__global__ void meant_kernel(const float* __restrict__ x, int L, int Cd, float* __restrict__ out, long n)
{
  long e = (long)blockIdx.x * blockDim.x + threadIdx.x;
  if (e >= n) return;
  int c = (int)(e % Cd); long b = e / Cd;
  float s = 0.f;
  for (int t = 0; t < L; ++t) s += x[((size_t)b * L + t) * Cd + c];
  out[e] = s / (float)L;
}

__global__ void pad_seasonal_kernel(const float* __restrict__ se, float* __restrict__ out,
                                    int Lh, int Lf, int Cd, long n)
{
  long e = (long)blockIdx.x * blockDim.x + threadIdx.x;
  if (e >= n) return;
  int c = (int)(e % Cd);
  long rt = e / Cd;
  int t = (int)(rt % Lf);
  long b = rt / Lf;
  out[e] = (t < Lh) ? se[((size_t)b * Lh + t) * Cd + c] : 0.f;
}

__global__ void trend_conv_kernel(const float* __restrict__ T, const float* __restrict__ Wt,
                                  float* __restrict__ rt, int L, long n)
{
  long e = (long)blockIdx.x * blockDim.x + threadIdx.x;
  if (e >= n) return;
  int t = (int)(e % L); long b = e / L;
  float acc = 0.f;
  for (int kk = 0; kk < 3; ++kk) {
    int srow = t + kk - 1; if (srow < 0) srow += L; if (srow >= L) srow -= L;
    const float* row = T + ((size_t)b * L + srow) * 512;
    for (int c = 0; c < 512; ++c) acc += row[c] * Wt[c * 3 + kk];
  }
  rt[e] = acc;
}

__global__ void proj_kernel(const float* __restrict__ X, const float* __restrict__ Wp,
                            const float* __restrict__ bp, float* __restrict__ out, long n)
{
  long e = (long)blockIdx.x * blockDim.x + threadIdx.x;
  if (e >= n) return;
  const float* row = X + (size_t)e * 512;
  float s = bp[0];
  for (int d = 0; d < 512; ++d) s += row[d] * Wp[d];
  out[e] = s;
}

__global__ void final_kernel(const float* __restrict__ meanx, const float* __restrict__ rt,
                             const float* __restrict__ seas, float* __restrict__ out, long n)
{
  long e = (long)blockIdx.x * blockDim.x + threadIdx.x;
  if (e >= n) return;
  int c = (int)(e % 21);
  long rj = e / 21;
  int j = (int)(rj % 256);
  long b = rj / 256;
  int t = 1024 + j;
  out[e] = meanx[b * 21 + c] + rt[b * 1280 + t] + seas[b * 1280 + t];
}

// ---------------------------------------------------------------- host orchestration
extern "C" void kernel_launch(void* const* d_in, const int* in_sizes, int n_in,
                              void* d_out, int out_size, void* d_ws, size_t ws_size,
                              hipStream_t stream) {
  (void)in_sizes; (void)n_in; (void)out_size; (void)ws_size;
  const long Bn = 32, LE = 1024, LDv = 1280, Dm = 512, FF = 2048, Cin = 21;
  const long RE = Bn * LE, RD = Bn * LDv;
  const int  TKE = 34, TKD = 35;

  auto in = [&](int i) -> const float* { return (const float*)d_in[i]; };
  // inputs 0..3: x_enc, x_dec, time, target_time ; params (jax tree order) follow:
  const float* x_enc  = in(0);
  const float* timeA  = in(2);
  const float* ttimeA = in(3);
  const int P = 4;
  // params: dec_emb, dec_layers[0]{W1,W2,cross{k,o,q,v}{W,b},self{...},trend},
  //         dec_norm{b,g}, dec_proj{W,b}, enc_emb, enc_layers[l]{W1,W2,attn{k,o,q,v}{W,b}},
  //         enc_norm{b,g}, time_emb
  const int iDecEmb = P + 0;
  const int iDW1 = P + 1, iDW2 = P + 2;
  const int iCkW = P + 3, iCkB = P + 4, iCoW = P + 5, iCoB = P + 6,
            iCqW = P + 7, iCqB = P + 8, iCvW = P + 9, iCvB = P + 10;
  const int iSkW = P + 11, iSkB = P + 12, iSoW = P + 13, iSoB = P + 14,
            iSqW = P + 15, iSqB = P + 16, iSvW = P + 17, iSvB = P + 18;
  const int iTrend = P + 19;
  const int iDnB = P + 20, iDnG = P + 21;
  const int iPW = P + 22, iPB = P + 23;
  const int iEncEmb = P + 24;
  const int iEL = P + 25;                          // enc layer l base: iEL + l*10
  const int iEnB = P + 45, iEnG = P + 46;
  const int iTemb = P + 47;

  // bump allocator over d_ws
  char* wsp = (char*)d_ws;
  size_t off = 0;
  auto alloc = [&](size_t bytes) -> void* {
    void* p = wsp + off;
    off += (bytes + 255) & ~(size_t)255;
    return p;
  };
  auto gb = [](long n) { return dim3((unsigned)((n + 255) / 256)); };

  auto toBf = [&](int idx, long n) -> bf16* {
    bf16* p = (bf16*)alloc((size_t)n * sizeof(bf16));
    f2bf_kernel<<<gb(n), 256, 0, stream>>>(in(idx), p, n);
    return p;
  };

  // weights -> bf16
  bf16 *eW[2][6];   // qW,kW,vW,oW,W1,W2 per enc layer
  for (int l = 0; l < 2; ++l) {
    int base = iEL + l * 10;
    eW[l][4] = toBf(base + 0, FF * Dm);       // W1
    eW[l][5] = toBf(base + 1, Dm * FF);       // W2
    eW[l][1] = toBf(base + 2, Dm * Dm);       // kW
    eW[l][3] = toBf(base + 4, Dm * Dm);       // oW
    eW[l][0] = toBf(base + 6, Dm * Dm);       // qW
    eW[l][2] = toBf(base + 8, Dm * Dm);       // vW
  }
  bf16* dW1b = toBf(iDW1, FF * Dm);
  bf16* dW2b = toBf(iDW2, Dm * FF);
  bf16* cWk = toBf(iCkW, Dm * Dm); bf16* cWo = toBf(iCoW, Dm * Dm);
  bf16* cWq = toBf(iCqW, Dm * Dm); bf16* cWv = toBf(iCvW, Dm * Dm);
  bf16* sWk = toBf(iSkW, Dm * Dm); bf16* sWo = toBf(iSoW, Dm * Dm);
  bf16* sWq = toBf(iSqW, Dm * Dm); bf16* sWv = toBf(iSvW, Dm * Dm);

  // activation buffers
  float* Xf   = (float*)alloc(RD * Dm * 4);  bf16* Xb   = (bf16*)alloc(RD * Dm * 2);
  float* Sf   = (float*)alloc(RD * Dm * 4);  bf16* Sb   = (bf16*)alloc(RD * Dm * 2);
  bf16*  Qb   = (bf16*) alloc(RD * Dm * 2);  bf16* Kb   = (bf16*)alloc(RD * Dm * 2);
  float* Vf   = (float*)alloc(RD * Dm * 4);
  float* Yf   = (float*)alloc(RD * Dm * 4);
  bf16*  Aggb = (bf16*) alloc(RD * Dm * 2);
  float* ENCf = (float*)alloc(RE * Dm * 4);  bf16* ENCb = (bf16*)alloc(RE * Dm * 2);
  float* Tacc = (float*)alloc(RD * Dm * 4);
  void*  arena = alloc((size_t)Bn * LDv * LDv * 4);   // corr (210MB) / FFN hidden (168MB)
  float* corrF = (float*)arena;
  bf16*  Hb    = (bf16*)arena;
  float* cm    = (float*)alloc(Bn * Dm * 4);
  float* mcorr = (float*)alloc(Bn * LDv * 4);
  int*   dly   = (int*)  alloc(Bn * 40 * 4);
  float* wts   = (float*)alloc(Bn * 40 * 4);
  float* se    = (float*)alloc(Bn * LE * Cin * 4);
  float* sfull = (float*)alloc(Bn * LDv * Cin * 4);
  float* meanx = (float*)alloc(Bn * Cin * 4);
  float* rt    = (float*)alloc(Bn * LDv * 4);
  float* seass = (float*)alloc(Bn * LDv * 4);

  auto gemm = [&](const bf16* A, long sAb, const bf16* W, long sWb, const float* bias,
                  const float* res, long sRb, float* Cf, long sCfb, bf16* Cb, long sCbb,
                  int M, int N, int K, int act, int batch) {
    dim3 g(N / 128, M / 128, batch);
    gemm_bf16_kernel<<<g, 256, 0, stream>>>(A, sAb, W, sWb, bias, res, sRb,
                                            Cf, sCfb, Cb, sCbb, M, N, K, act);
  };

  // ============================ ENCODER ============================
  conv_emb_kernel<<<(unsigned)RE, 256, 0, stream>>>(
      x_enc, in(iEncEmb), timeA, timeA, (int)LE, in(iTemb), Xf, Xb, (int)LE, (int)Cin);

  for (int l = 0; l < 2; ++l) {
    int base = iEL + l * 10;
    const float *qb = in(base+7), *kb = in(base+3), *vb = in(base+9), *ob = in(base+5);
    gemm(Xb,0, eW[l][0],0, qb, nullptr,0, nullptr,0, Qb,0, (int)RE,512,512,0,1);
    gemm(Xb,0, eW[l][1],0, kb, nullptr,0, nullptr,0, Kb,0, (int)RE,512,512,0,1);
    gemm(Xb,0, eW[l][2],0, vb, nullptr,0, Vf,0, nullptr,0, (int)RE,512,512,0,1);
    gemm(Qb, LE*Dm, Kb, LE*Dm, nullptr, nullptr,0, corrF, LE*LE, nullptr,0,
         (int)LE,(int)LE,512,0,(int)Bn);
    corr_mean_kernel<<<dim3((unsigned)LE,(unsigned)Bn),256,0,stream>>>(corrF,(int)LE,mcorr,1.f/512.f);
    topk_softmax_kernel<<<(unsigned)Bn,256,0,stream>>>(mcorr,(int)LE,TKE,dly,wts);
    agg_kernel<<<dim3(gb(LE*Dm).x,(unsigned)Bn),256,0,stream>>>(Vf,dly,wts,TKE,Aggb,(int)LE,LE*Dm);
    gemm(Aggb,0, eW[l][3],0, ob, Xf,0, Yf,0, nullptr,0, (int)RE,512,512,0,1);
    decomp_kernel<<<gb(RE*Dm),256,0,stream>>>(Yf,(int)LE,512,Sf,Sb,nullptr,RE*Dm);
    gemm(Sb,0, eW[l][4],0, nullptr, nullptr,0, nullptr,0, Hb,0, (int)RE,(int)FF,512,1,1);
    gemm(Hb,0, eW[l][5],0, nullptr, Sf,0, Yf,0, nullptr,0, (int)RE,512,(int)FF,0,1);
    decomp_kernel<<<gb(RE*Dm),256,0,stream>>>(Yf,(int)LE,512,Xf,Xb,nullptr,RE*Dm);
  }
  ln_row_kernel<<<(unsigned)RE,256,0,stream>>>(Xf, in(iEnG), in(iEnB), Yf);
  colmean_kernel<<<gb(Bn*Dm),256,0,stream>>>(Yf,(int)LE,cm,Bn*Dm);
  colsub_kernel<<<gb(RE*Dm),256,0,stream>>>(Yf,cm,ENCf,ENCb,(int)LE,RE*Dm);

  // ============================ DECODER ============================
  decomp_kernel<<<gb(Bn*LE*Cin),256,0,stream>>>(x_enc,(int)LE,(int)Cin,se,nullptr,nullptr,Bn*LE*Cin);
  meant_kernel<<<gb(Bn*Cin),256,0,stream>>>(x_enc,(int)LE,(int)Cin,meanx,Bn*Cin);
  pad_seasonal_kernel<<<gb(Bn*LDv*Cin),256,0,stream>>>(se,sfull,(int)LE,(int)LDv,(int)Cin,Bn*LDv*Cin);
  conv_emb_kernel<<<(unsigned)RD,256,0,stream>>>(
      sfull, in(iDecEmb), timeA, ttimeA, (int)LE, in(iTemb), Xf, Xb, (int)LDv, (int)Cin);
  fill_f32_kernel<<<gb(RD*Dm),256,0,stream>>>(Tacc,0.f,RD*Dm);

  // ---- self attention (L = 1280)
  gemm(Xb,0, sWq,0, in(iSqB), nullptr,0, nullptr,0, Qb,0, (int)RD,512,512,0,1);
  gemm(Xb,0, sWk,0, in(iSkB), nullptr,0, nullptr,0, Kb,0, (int)RD,512,512,0,1);
  gemm(Xb,0, sWv,0, in(iSvB), nullptr,0, Vf,0, nullptr,0, (int)RD,512,512,0,1);
  gemm(Qb, LDv*Dm, Kb, LDv*Dm, nullptr, nullptr,0, corrF, LDv*LDv, nullptr,0,
       (int)LDv,(int)LDv,512,0,(int)Bn);
  corr_mean_kernel<<<dim3((unsigned)LDv,(unsigned)Bn),256,0,stream>>>(corrF,(int)LDv,mcorr,1.f/512.f);
  topk_softmax_kernel<<<(unsigned)Bn,256,0,stream>>>(mcorr,(int)LDv,TKD,dly,wts);
  agg_kernel<<<dim3(gb(LDv*Dm).x,(unsigned)Bn),256,0,stream>>>(Vf,dly,wts,TKD,Aggb,(int)LDv,LDv*Dm);
  gemm(Aggb,0, sWo,0, in(iSoB), Xf,0, Yf,0, nullptr,0, (int)RD,512,512,0,1);
  decomp_kernel<<<gb(RD*Dm),256,0,stream>>>(Yf,(int)LDv,512,Sf,Sb,Tacc,RD*Dm);

  // ---- cross attention (q len 1280, k/v from encoder, zero-padded to 1280)
  gemm(Sb,0, cWq,0, in(iCqB), nullptr,0, nullptr,0, Qb,0, (int)RD,512,512,0,1);
  fill_u16_kernel<<<gb(RD*Dm),256,0,stream>>>((unsigned short*)Kb,RD*Dm);
  fill_f32_kernel<<<gb(RD*Dm),256,0,stream>>>(Vf,0.f,RD*Dm);
  gemm(ENCb, LE*Dm, cWk,0, in(iCkB), nullptr,0, nullptr,0, Kb, LDv*Dm,
       (int)LE,512,512,0,(int)Bn);
  gemm(ENCb, LE*Dm, cWv,0, in(iCvB), nullptr,0, Vf, LDv*Dm, nullptr,0,
       (int)LE,512,512,0,(int)Bn);
  gemm(Qb, LDv*Dm, Kb, LDv*Dm, nullptr, nullptr,0, corrF, LDv*LDv, nullptr,0,
       (int)LDv,(int)LDv,512,0,(int)Bn);
  corr_mean_kernel<<<dim3((unsigned)LDv,(unsigned)Bn),256,0,stream>>>(corrF,(int)LDv,mcorr,1.f/512.f);
  topk_softmax_kernel<<<(unsigned)Bn,256,0,stream>>>(mcorr,(int)LDv,TKD,dly,wts);
  agg_kernel<<<dim3(gb(LDv*Dm).x,(unsigned)Bn),256,0,stream>>>(Vf,dly,wts,TKD,Aggb,(int)LDv,LDv*Dm);
  gemm(Aggb,0, cWo,0, in(iCoB), Sf,0, Yf,0, nullptr,0, (int)RD,512,512,0,1);
  decomp_kernel<<<gb(RD*Dm),256,0,stream>>>(Yf,(int)LDv,512,Xf,Xb,Tacc,RD*Dm);

  // ---- FFN
  gemm(Xb,0, dW1b,0, nullptr, nullptr,0, nullptr,0, Hb,0, (int)RD,(int)FF,512,1,1);
  gemm(Hb,0, dW2b,0, nullptr, Xf,0, Yf,0, nullptr,0, (int)RD,512,(int)FF,0,1);
  decomp_kernel<<<gb(RD*Dm),256,0,stream>>>(Yf,(int)LDv,512,Sf,nullptr,Tacc,RD*Dm);

  // ---- trend conv, final norm, projection, output
  trend_conv_kernel<<<gb(Bn*LDv),256,0,stream>>>(Tacc,in(iTrend),rt,(int)LDv,Bn*LDv);
  ln_row_kernel<<<(unsigned)RD,256,0,stream>>>(Sf, in(iDnG), in(iDnB), Yf);
  colmean_kernel<<<gb(Bn*Dm),256,0,stream>>>(Yf,(int)LDv,cm,Bn*Dm);
  colsub_kernel<<<gb(RD*Dm),256,0,stream>>>(Yf,cm,Xf,nullptr,(int)LDv,RD*Dm);
  proj_kernel<<<gb(Bn*LDv),256,0,stream>>>(Xf,in(iPW),in(iPB),seass,Bn*LDv);
  final_kernel<<<gb(Bn*256*Cin),256,0,stream>>>(meanx,rt,seass,(float*)d_out,Bn*256*Cin);
}
